// NaiveFourierKANLayer_1254130450856
// MI455X (gfx1250) — matrix operand order repaired and checked by reading, not compiled, not run on hardware
//
#include <hip/hip_runtime.h>

// ---------------------------------------------------------------------------
// FourierKAN forward as a fused f16-WMMA GEMM (M=8192, N=256, K=32768).
//   y[b,o] = sum_{i,g} cos(x[b,i]*g)*W0[o,i,g] + sin(x[b,i]*g)*W1[o,i,g] + bias
// Harmonics generated with the angle-addition recurrence (2 transcendentals
// per 128 activation values); weights pre-cast to f16 and kept L2-resident.
// Block = 256 threads (8 waves), tile = 32 rows x all 256 cols, double-
// buffered activation LDS so act-gen VALU co-executes with WMMA across waves.
// ---------------------------------------------------------------------------

typedef _Float16 v16h __attribute__((ext_vector_type(16)));
typedef _Float16 v8h  __attribute__((ext_vector_type(8)));
typedef _Float16 h4   __attribute__((ext_vector_type(4)));
typedef float    v8f  __attribute__((ext_vector_type(8)));
typedef float    v4f  __attribute__((ext_vector_type(4)));

#define BDIM   8192
#define IDIM   256
#define ODIM   256
#define GRID   64
#define KTOT   (IDIM * GRID * 2)   // 32768
#define ROWS   32                  // M-tile rows per block (B-fragment reuse x2)
#define ICH    8                   // inputs per K-chunk
#define KC     (ICH * 128)         // 1024 K-values per chunk
#define NCHUNK (KTOT / KC)         // 32
#define KSTEPS (KC / 32)           // 32 wmma k-steps per chunk
#define LDSW   1032                // KC + 8 halfs pad: dword stride %64 == 4

// --- Kernel 1: repack fouriercoeffs (2,O,I,G) f32 -> W16[o][k] f16,
//     k = i*128 + s*64 + (g-1). Dest linear index == o*32768 + i*128 + s*64 + g0.
__global__ __launch_bounds__(256) void repack_weights_f16(
    const float* __restrict__ fc, _Float16* __restrict__ w16)
{
    int e0 = (blockIdx.x * 256 + threadIdx.x) * 4;   // 4 contiguous g per thread
    int g0 = e0 & 63;
    int s  = (e0 >> 6) & 1;
    int i  = (e0 >> 7) & 255;
    int o  = e0 >> 15;
    size_t src = (((size_t)(s * ODIM + o)) * IDIM + i) * GRID + g0;
    v4f v = *(const v4f*)(fc + src);
    h4 hv = { (_Float16)v.x, (_Float16)v.y, (_Float16)v.z, (_Float16)v.w };
    *(h4*)(w16 + e0) = hv;
}

// Generate one (row, input) pair's 64 cos + 64 sin f16 harmonics into LDS.
__device__ __forceinline__ void gen_act(_Float16* __restrict__ buf,
                                        const float* __restrict__ x,
                                        int row0, int m, int il, int chunk)
{
    float xv = x[(size_t)(row0 + m) * IDIM + chunk * ICH + il];
    float c1, s1;
    __sincosf(xv, &s1, &c1);              // 2 transcendentals per 128 values
    float cg = c1, sg = s1;               // g = 1
    const int base = m * LDSW + il * 128;
    #pragma unroll
    for (int q = 0; q < 16; ++q) {        // emit 4 harmonics per iteration
        float ca = cg,            sa = sg;
        float cb = ca*c1 - sa*s1, sb = sa*c1 + ca*s1;
        float cc = cb*c1 - sb*s1, sc = sb*c1 + cb*s1;
        float cd = cc*c1 - sc*s1, sd = sc*c1 + cc*s1;
        h4 cp = { (_Float16)ca, (_Float16)cb, (_Float16)cc, (_Float16)cd };
        h4 sp = { (_Float16)sa, (_Float16)sb, (_Float16)sc, (_Float16)sd };
        *(h4*)(&buf[base + 4*q])      = cp;   // cos block: k = il*128 + g-1
        *(h4*)(&buf[base + 64 + 4*q]) = sp;   // sin block: k = il*128 + 64 + g-1
        cg = cd*c1 - sd*s1;
        sg = sd*c1 + cd*s1;
    }
}

// --- Kernel 2: fused activation-gen + WMMA GEMM.
__global__ __launch_bounds__(256) void fourier_kan_wmma(
    const float* __restrict__ x, const _Float16* __restrict__ w16,
    const float* __restrict__ bias, float* __restrict__ out)
{
    __shared__ _Float16 act[2][ROWS * LDSW];   // double-buffered, ~132 KB

    const int tid  = threadIdx.x;
    const int lane = tid & 31;
    const int wave = tid >> 5;      // 0..7
    const int lo   = lane & 15;
    const int hi   = lane >> 4;     // 0 or 1
    const int row0 = blockIdx.x * ROWS;

    // activation-phase mapping: m = lane, il = wave (32 rows x 8 inputs/chunk)
    const int m_act = tid & 31;
    const int il    = tid >> 5;

    v8f acc00 = {}, acc01 = {}, acc10 = {}, acc11 = {};
    const int n0 = wave * 32 + lo;  // column this lane owns, N-tile 0
    const int n1 = n0 + 16;         // N-tile 1

    gen_act(&act[0][0], x, row0, m_act, il, 0);
    __syncthreads();

    for (int c = 0; c < NCHUNK; ++c) {
        // generate next chunk into the other buffer (overlaps with MMA
        // of current chunk across skewed waves on the same SIMD)
        if (c + 1 < NCHUNK)
            gen_act(&act[(c + 1) & 1][0], x, row0, m_act, il, c + 1);

        // ---- MMA over current chunk ----
        const _Float16* __restrict__ ab = &act[c & 1][0];
        // B fragment: lane<16 -> col n, K k..k+15 ; lane>=16 -> K k+16..k+31
        const _Float16* wb0 = w16 + (size_t)n0 * KTOT + c * KC + hi * 16;
        const _Float16* wb1 = w16 + (size_t)n1 * KTOT + c * KC + hi * 16;

        #pragma unroll 2
        for (int kk = 0; kk < KSTEPS; ++kk) {
            const int lk0 = kk * 32;
            // A contiguous K to match B: lanes 0-15 K 0..15, lanes 16-31
            // K 16..31; M-tile0 rows lo, M-tile1 rows lo+16.
            const int offA0 = lo * LDSW + lk0 + hi * 16;
            const int offA1 = offA0 + 16 * LDSW;

            v8h a0l = *(const v8h*)(&ab[offA0]);
            v8h a0h = *(const v8h*)(&ab[offA0 + 8]);
            v16h a0 = __builtin_shufflevector(a0l, a0h,
                      0,1,2,3,4,5,6,7,8,9,10,11,12,13,14,15);
            v8h a1l = *(const v8h*)(&ab[offA1]);
            v8h a1h = *(const v8h*)(&ab[offA1 + 8]);
            v16h a1 = __builtin_shufflevector(a1l, a1h,
                      0,1,2,3,4,5,6,7,8,9,10,11,12,13,14,15);

            v16h b0 = *(const v16h*)(wb0 + lk0);   // 32B-aligned global loads
            v16h b1 = *(const v16h*)(wb1 + lk0);

            acc00 = __builtin_amdgcn_wmma_f32_16x16x32_f16(
                        false, a0, false, b0, (short)0, acc00, false, false);
            acc01 = __builtin_amdgcn_wmma_f32_16x16x32_f16(
                        false, a0, false, b1, (short)0, acc01, false, false);
            acc10 = __builtin_amdgcn_wmma_f32_16x16x32_f16(
                        false, a1, false, b0, (short)0, acc10, false, false);
            acc11 = __builtin_amdgcn_wmma_f32_16x16x32_f16(
                        false, a1, false, b1, (short)0, acc11, false, false);
        }
        __syncthreads();   // one barrier per chunk (double-buffered)
    }

    // ---- epilogue: C layout VGPR j -> (M = j + hi*8, N = lane%16) ----
    const float bv0 = bias[n0];
    const float bv1 = bias[n1];
    #pragma unroll
    for (int j = 0; j < 8; ++j) {
        const int m = j + hi * 8;
        out[(size_t)(row0 + m)      * ODIM + n0] = acc00[j] + bv0;
        out[(size_t)(row0 + m)      * ODIM + n1] = acc01[j] + bv1;
        out[(size_t)(row0 + 16 + m) * ODIM + n0] = acc10[j] + bv0;
        out[(size_t)(row0 + 16 + m) * ODIM + n1] = acc11[j] + bv1;
    }
}

extern "C" void kernel_launch(void* const* d_in, const int* in_sizes, int n_in,
                              void* d_out, int out_size, void* d_ws, size_t ws_size,
                              hipStream_t stream)
{
    const float* x    = (const float*)d_in[0];   // (8192, 256)
    const float* fc   = (const float*)d_in[1];   // (2, 256, 256, 64)
    const float* bias = (const float*)d_in[2];   // (1, 256)
    float* out        = (float*)d_out;           // (8192, 256)

    _Float16* w16 = (_Float16*)d_ws;             // 256*32768*2B = 16.8 MB

    // 1) repack weights to f16 [o][k]
    repack_weights_f16<<<(ODIM * KTOT) / (4 * 256), 256, 0, stream>>>(fc, w16);

    // 2) fused FourierKAN GEMM: 256 blocks of 32 rows each
    fourier_kan_wmma<<<BDIM / ROWS, 256, 0, stream>>>(x, w16, bias, out);
}
